// BaseQVLayer_27058293964889
// MI455X (gfx1250) — matrix operand order repaired and checked by reading, not compiled, not run on hardware
//
#include <hip/hip_runtime.h>
#include <hip/hip_bf16.h>

#define NXX 8192
#define NYY 8192
#define FDIM 1024
#define EMB 512

typedef __bf16 bf16;
typedef bf16 bf16x8  __attribute__((ext_vector_type(8)));
typedef bf16 bf16x16 __attribute__((ext_vector_type(16)));
typedef float f32x4  __attribute__((ext_vector_type(4)));
typedef float f32x8  __attribute__((ext_vector_type(8)));
typedef int   i32x4  __attribute__((ext_vector_type(4)));
typedef int   i32x8  __attribute__((ext_vector_type(8)));

// ---- WMMA fragment loaders (wave32; contiguous k = 16*hi + s, same order in A and B) -------------
// A (16xK slice, row-major source): lane n=l&15 -> row n.
// lo-lanes: K = k0+[0..15]; hi-lanes: K = k0+[16..31].  Two 16B loads.
__device__ __forceinline__ bf16x16 load_a_frag(const bf16* row_base, int k0, int hi) {
  const bf16* p = row_base + k0 + (hi ? 16 : 0);
  bf16x8 lo = *(const bf16x8*)(p);
  bf16x8 hh = *(const bf16x8*)(p + 8);
  bf16x16 r;
#pragma unroll
  for (int i = 0; i < 8; ++i) { r[i] = lo[i]; r[i + 8] = hh[i]; }
  return r;
}

// B (Kx16 slice, source laid out column-contiguous: src[n*ldk + k]):
// lane n=l&15 -> col n; lo-lanes K=k0+[0..15], hi-lanes K=k0+[16..31]. One 32B load.
__device__ __forceinline__ bf16x16 load_b_frag(const bf16* col_base, int k0, int hi) {
  return *(const bf16x16*)(col_base + k0 + (hi ? 16 : 0));
}

#define WMMA_BF16(a, b, c) \
  __builtin_amdgcn_wmma_f32_16x16x32_bf16(false, (a), false, (b), (short)0, (c), false, false)

// ---- Projection: P = X @ W + bias  (f32 LDS-tiled; <6% of total FLOPs) ----
__global__ __launch_bounds__(256) void proj_kernel(
    const float* __restrict__ X, const float* __restrict__ W,
    const float* __restrict__ bias, bf16* __restrict__ P,
    bf16* __restrict__ PT, int Nrows) {
  __shared__ float Xs[64][17];
  __shared__ float Ws[16][65];
  const int tid = threadIdx.x;
  const int tx = tid & 15, ty = tid >> 4;
  const int m0 = blockIdx.x * 64, n0 = blockIdx.y * 64;
  float c[4][4] = {};
  for (int kt = 0; kt < FDIM; kt += 16) {
#pragma unroll
    for (int r = 0; r < 4; ++r) {
      int idx = tid + r * 256;
      Xs[idx >> 4][idx & 15] = X[(size_t)(m0 + (idx >> 4)) * FDIM + kt + (idx & 15)];
      Ws[idx >> 6][idx & 63] = W[(size_t)(kt + (idx >> 6)) * EMB + n0 + (idx & 63)];
    }
    __syncthreads();
#pragma unroll
    for (int k = 0; k < 16; ++k)
#pragma unroll
      for (int i = 0; i < 4; ++i)
#pragma unroll
        for (int j = 0; j < 4; ++j)
          c[i][j] += Xs[ty * 4 + i][k] * Ws[k][tx * 4 + j];
    __syncthreads();
  }
#pragma unroll
  for (int i = 0; i < 4; ++i)
#pragma unroll
    for (int j = 0; j < 4; ++j) {
      int m = m0 + ty * 4 + i, nn = n0 + tx * 4 + j;
      float v = c[i][j] + bias[nn];
      P[(size_t)m * EMB + nn] = (bf16)v;
      if (PT) PT[(size_t)nn * Nrows + m] = (bf16)v;
    }
}

// ---- Row squared-norms: D[i] = sum_e P[i,e]^2  (wave32 shuffle reduce) ----
__global__ __launch_bounds__(256) void dsum_kernel(const bf16* __restrict__ P,
                                                   float* __restrict__ D) {
  const int w = threadIdx.x >> 5, lane = threadIdx.x & 31;
  const int row = blockIdx.x * 8 + w;
  const bf16* p = P + (size_t)row * EMB + lane * 16;
  bf16x8 a = *(const bf16x8*)p;
  bf16x8 b = *(const bf16x8*)(p + 8);
  float s = 0.f;
#pragma unroll
  for (int i = 0; i < 8; ++i) {
    float fa = (float)a[i], fb = (float)b[i];
    s += fa * fa + fb * fb;
  }
#pragma unroll
  for (int off = 16; off > 0; off >>= 1) s += __shfl_xor(s, off, 32);
  if (lane == 0) D[row] = s;
}

// ---- Transpose + bf16-convert Wg ------------------------------------------
__global__ __launch_bounds__(256) void wgt_kernel(const float* __restrict__ Wg,
                                                  bf16* __restrict__ WgT) {
  int idx = blockIdx.x * 256 + threadIdx.x;
  int e = idx >> 9, o = idx & 511;
  WgT[(size_t)o * EMB + e] = (bf16)Wg[idx];
}

// ---- Fused: gwf = (2*(xp@yp.T)/(Dc+Dr)).T @ xp  (A never materialized) ----
// Block: 32 y-rows (j) x full EMB (512) of gwf, 8 waves (wave w owns e=[64w,64w+64)).
// yp tile is DMA'd once into padded LDS by the Tensor Data Mover (TDM); the D#
// pad feature (8 dwords every 256 dwords) realizes the Ys[32][512+16] layout.
// Loop over x in 128-row chunks: each wave computes its 16x32 slab of S via WMMA,
// scales, stages S^T in LDS (bf16), then all waves run the second WMMA chain.
__global__ __launch_bounds__(256) void fused_affinity_kernel(
    const bf16* __restrict__ xp, const bf16* __restrict__ xpT,
    const bf16* __restrict__ yp, const float* __restrict__ Dcol,
    const float* __restrict__ Drow, bf16* __restrict__ gwf) {
  __shared__ bf16 Ys[32][EMB + 16];  // yp tile, +16 pad (1056B rows; 32B-aligned)
  __shared__ bf16 St[32][136];       // S^T tile [j][i], 272B rows (16B-aligned)
  const int tid = threadIdx.x;
  const int w = tid >> 5, lane = tid & 31;
  const int n = lane & 15, hi = lane >> 4;
  const int j0 = blockIdx.x * 32;
  const int e0 = w * 64;

  // ---- TDM: DMA yp[j0..j0+32][0..512) -> Ys with hardware row padding ----
  if (w == 0) {
    unsigned lds_base = (unsigned)(uintptr_t)(&Ys[0][0]);
    unsigned long long ga = (unsigned long long)(uintptr_t)(yp + (size_t)j0 * EMB);
    i32x4 g0;
    g0[0] = 1;                                            // count=1 (valid user D#)
    g0[1] = (int)lds_base;                                // lds_addr
    g0[2] = (int)(ga & 0xffffffffu);                      // global_addr[31:0]
    g0[3] = (int)((ga >> 32) & 0x1ffffffu) | (int)(2u << 30);  // ga[56:32] | type=2
    i32x8 g1;
    g1[0] = (1 << 16)      // data_size = 2 bytes
          | (1 << 20)      // pad_enable
          | (7 << 22)      // pad_interval: 256 dwords (= one 512-elem row)
          | (7 << 25);     // pad_amount: 8 dwords (= 16 bf16)
    g1[1] = (EMB & 0xffff) << 16;        // tensor_dim0[15:0]=512
    g1[2] = (EMB >> 16) | (32 << 16);    // tensor_dim0[31:16] | tensor_dim1=32
    g1[3] = (EMB << 16);                 // tensor_dim1[31:16]=0 | tile_dim0=512
    g1[4] = 32;                          // tile_dim1=32, tile_dim2=0
    g1[5] = EMB;                         // tensor_dim0_stride[31:0]=512
    g1[6] = 0;                           // stride0[47:32]=0, dim1_stride lo=0
    g1[7] = 0;
    asm volatile("tensor_load_to_lds %0, %1" :: "s"(g0), "s"(g1) : "memory");
    __builtin_amdgcn_s_wait_tensorcnt(0);
  }
  __syncthreads();

  f32x8 acc[2][4] = {};  // gwf accumulators: [j-tile][e-tile]
  const float dr0 = Drow[j0 + n];
  const float dr1 = Drow[j0 + 16 + n];

  for (int i0 = 0; i0 < NXX; i0 += 128) {
    // ---- matmul-1: S[iw..iw+16, j0..j0+32] = xp @ yp.T (B from LDS) ----
    const int iw = i0 + w * 16;
    f32x8 s0 = {}, s1 = {};
    const bf16* arow  = xp + (size_t)(iw + n) * EMB;
    const bf16* bcol0 = &Ys[n][0];
    const bf16* bcol1 = &Ys[16 + n][0];
    if (i0 + 128 < NXX)  // prefetch next chunk's A rows (global_prefetch_b8)
      __builtin_prefetch(xp + (size_t)(iw + 128 + n) * EMB, 0, 3);
#pragma unroll 4
    for (int k0 = 0; k0 < EMB; k0 += 32) {
      bf16x16 a  = load_a_frag(arow, k0, hi);
      bf16x16 b0 = load_b_frag(bcol0, k0, hi);
      bf16x16 b1 = load_b_frag(bcol1, k0, hi);
      s0 = WMMA_BF16(a, b0, s0);
      s1 = WMMA_BF16(a, b1, s1);
    }
    // ---- scale: 2/(||x_i||^2 + ||y_j||^2), stage S^T in LDS as bf16 ----
    f32x4 dclo = *(const f32x4*)(Dcol + iw + hi * 8);
    f32x4 dchi = *(const f32x4*)(Dcol + iw + hi * 8 + 4);
    bf16x8 p0, p1;
#pragma unroll
    for (int r = 0; r < 8; ++r) {
      float dc = (r < 4) ? dclo[r] : dchi[r - 4];
      p0[r] = (bf16)(2.0f * s0[r] * __builtin_amdgcn_rcpf(dc + dr0));
      p1[r] = (bf16)(2.0f * s1[r] * __builtin_amdgcn_rcpf(dc + dr1));
    }
    // C-layout element (m = r + 8*hi, col n) -> St[n][i_local], 16B contiguous store
    *(bf16x8*)&St[n][w * 16 + hi * 8]      = p0;
    *(bf16x8*)&St[16 + n][w * 16 + hi * 8] = p1;
    __syncthreads();
    // ---- matmul-2: acc[j, e] += S^T @ xp  (A = St from LDS, B = xpT) ----
#pragma unroll
    for (int kk = 0; kk < 128; kk += 32) {
      bf16x16 a0 = load_a_frag(&St[n][0], kk, hi);
      bf16x16 a1 = load_a_frag(&St[16 + n][0], kk, hi);
#pragma unroll
      for (int et = 0; et < 4; ++et) {
        const bf16* bcol = xpT + (size_t)(e0 + et * 16 + n) * NXX + i0;
        bf16x16 b = load_b_frag(bcol, kk, hi);
        acc[0][et] = WMMA_BF16(a0, b, acc[0][et]);
        acc[1][et] = WMMA_BF16(a1, b, acc[1][et]);
      }
    }
    __syncthreads();
  }
  // ---- epilogue: write gwf tile (bf16, feeds the final WMMA GEMM) ----
#pragma unroll
  for (int jt = 0; jt < 2; ++jt)
#pragma unroll
    for (int et = 0; et < 4; ++et) {
      int col = e0 + et * 16 + n;
#pragma unroll
      for (int r = 0; r < 8; ++r) {
        int row = j0 + jt * 16 + hi * 8 + r;
        gwf[(size_t)row * EMB + col] = (bf16)acc[jt][et][r];
      }
    }
}

// ---- out = relu(gwf @ Wg + bg), M=8192 N=512 K=512, bf16 WMMA --------------
__global__ __launch_bounds__(256) void out_gemm_kernel(
    const bf16* __restrict__ gwf, const bf16* __restrict__ WgT,
    const float* __restrict__ bg, float* __restrict__ out) {
  const int tid = threadIdx.x;
  const int w = tid >> 5, lane = tid & 31;
  const int n = lane & 15, hi = lane >> 4;
  const int m0 = blockIdx.x * 128 + w * 16;
  const int n0 = blockIdx.y * 64;
  f32x8 acc[4] = {};
  const bf16* arow = gwf + (size_t)(m0 + n) * EMB;
#pragma unroll 4
  for (int k0 = 0; k0 < EMB; k0 += 32) {
    bf16x16 a = load_a_frag(arow, k0, hi);
#pragma unroll
    for (int nt = 0; nt < 4; ++nt) {
      const bf16* bcol = WgT + (size_t)(n0 + nt * 16 + n) * EMB;
      bf16x16 b = load_b_frag(bcol, k0, hi);
      acc[nt] = WMMA_BF16(a, b, acc[nt]);
    }
  }
#pragma unroll
  for (int nt = 0; nt < 4; ++nt) {
    int col = n0 + nt * 16 + n;
    float bv = bg[col];
#pragma unroll
    for (int r = 0; r < 8; ++r) {
      float v = acc[nt][r] + bv;
      out[(size_t)(m0 + hi * 8 + r) * EMB + col] = v > 0.f ? v : 0.f;
    }
  }
}

extern "C" void kernel_launch(void* const* d_in, const int* in_sizes, int n_in,
                              void* d_out, int out_size, void* d_ws, size_t ws_size,
                              hipStream_t stream) {
  const float* x  = (const float*)d_in[0];
  const float* y  = (const float*)d_in[1];
  const float* Wx = (const float*)d_in[2];
  const float* bx = (const float*)d_in[3];
  const float* Wy = (const float*)d_in[4];
  const float* by = (const float*)d_in[5];
  const float* Wg = (const float*)d_in[6];
  const float* bg = (const float*)d_in[7];
  float* out = (float*)d_out;

  char* ws = (char*)d_ws;
  size_t off = 0;
  auto alloc = [&](size_t bytes) -> void* {
    void* p = ws + off;
    off += (bytes + 255) & ~(size_t)255;
    return p;
  };
  bf16* xp  = (bf16*)alloc((size_t)NXX * EMB * 2);
  bf16* yp  = (bf16*)alloc((size_t)NYY * EMB * 2);
  bf16* xpT = (bf16*)alloc((size_t)EMB * NXX * 2);
  bf16* gwf = (bf16*)alloc((size_t)NYY * EMB * 2);
  bf16* WgT = (bf16*)alloc((size_t)EMB * EMB * 2);
  float* Dc = (float*)alloc((size_t)NXX * 4);
  float* Dr = (float*)alloc((size_t)NYY * 4);

  proj_kernel<<<dim3(NXX / 64, EMB / 64), 256, 0, stream>>>(x, Wx, bx, xp, xpT, NXX);
  proj_kernel<<<dim3(NYY / 64, EMB / 64), 256, 0, stream>>>(y, Wy, by, yp, nullptr, NYY);
  dsum_kernel<<<NXX / 8, 256, 0, stream>>>(xp, Dc);
  dsum_kernel<<<NYY / 8, 256, 0, stream>>>(yp, Dr);
  wgt_kernel<<<(EMB * EMB) / 256, 256, 0, stream>>>(Wg, WgT);
  fused_affinity_kernel<<<NYY / 32, 256, 0, stream>>>(xp, xpT, yp, Dc, Dr, gwf);
  out_gemm_kernel<<<dim3(NYY / 128, EMB / 64), 256, 0, stream>>>(gwf, WgT, bg, out);
}
